// LearnedSegmentEncoder_28939489640461
// MI455X (gfx1250) — matrix operand order repaired and checked by reading, not compiled, not run on hardware
//
#include <hip/hip_runtime.h>
#include <hip/hip_bf16.h>

// ---- problem constants ----
#define BN 8
#define CC 128
#define DD 64
#define HH 256
#define WW 256
#define PP (HH * WW)
#define SS 32

// ---- tiling ----
#define TP 128            // pixels staged per block iteration
#define CHUNK 1024        // pixels per block
#define BLK_PER_B (PP / CHUNK)   // 64
#define XSTRIDE 136       // padded ushort stride for Xb rows (272B, 16B-aligned)

typedef __attribute__((ext_vector_type(16))) __bf16 v16bf;
typedef __attribute__((ext_vector_type(8)))  float  v8f;
typedef __attribute__((ext_vector_type(4)))  unsigned int u32x4;

union Frag { v16bf v; u32x4 q[2]; };

__device__ __forceinline__ unsigned short f2bf(float f) {
    unsigned int u = __float_as_uint(f);
    u += 0x7FFFu + ((u >> 16) & 1u);      // round-to-nearest-even
    return (unsigned short)(u >> 16);
}

__global__ void seg_init_kernel(float* ws, int n) {
    int i = blockIdx.x * blockDim.x + threadIdx.x;
    if (i < n) ws[i] = 0.0f;
}

__global__ __launch_bounds__(256)
void seg_fused_kernel(const int* __restrict__ labels,
                      const float* __restrict__ features,
                      const float* __restrict__ c1w, const float* __restrict__ c1b,
                      const float* __restrict__ c2w, const float* __restrict__ c2b,
                      float* __restrict__ gsum, float* __restrict__ gcnt) {
    __shared__ __align__(16) unsigned short Xb[TP][XSTRIDE];   // [pixel][chan] bf16
    __shared__ __align__(16) unsigned short W1s[DD][CC];       // [out][chan]  bf16
    __shared__ __align__(16) unsigned short W2s[DD][DD];       // [out2][dim]  bf16
    __shared__ __align__(16) unsigned short H1s[8][16][DD];    // per-wave [p][d] bf16
    __shared__ float segsum[SS][DD];
    __shared__ float segcnt[SS];
    __shared__ int   lbl[TP];

    const int b     = blockIdx.x / BLK_PER_B;
    const int chunk = blockIdx.x % BLK_PER_B;
    const int tid   = threadIdx.x;
    const int wv    = tid >> 5;
    const int lane  = tid & 31;
    const int n     = lane & 15;        // WMMA column within tile
    const int hi    = lane >> 4;        // lane-half select
    const int khalfA = hi ? 8 : 0;      // A-fragment K base offset
    const int khB    = hi ? 8 : 0;      // B-fragment K base offset

    // one-time per block: bf16 weights + zero accumulators
    for (int i = tid; i < DD * CC; i += 256) W1s[i / CC][i % CC] = f2bf(c1w[i]);
    for (int i = tid; i < DD * DD; i += 256) W2s[i / DD][i % DD] = f2bf(c2w[i]);
    for (int i = tid; i < SS * DD; i += 256) segsum[i / DD][i % DD] = 0.0f;
    if (tid < SS) segcnt[tid] = 0.0f;
    __syncthreads();

    const float* featb = features + (size_t)b * CC * PP;
    const int*   lblb  = labels   + (size_t)b * PP;

    for (int it = 0; it < CHUNK / TP; ++it) {
        const int pbase = chunk * CHUNK + it * TP;

        if (tid < TP) lbl[tid] = lblb[pbase + tid];

        // cooperative load: features (coalesced along p) -> bf16 transposed [p][c]
        for (int i = tid; i < CC * (TP / 2); i += 256) {
            int c  = i / (TP / 2);
            int pp = (i % (TP / 2)) * 2;
            float2 f = *(const float2*)(featb + (size_t)c * PP + pbase + pp);
            Xb[pp    ][c] = f2bf(f.x);
            Xb[pp + 1][c] = f2bf(f.y);
        }
        // prefetch next tile (global_prefetch_b8)
        if (it + 1 < CHUNK / TP) {
            int c = tid & (CC - 1);
            __builtin_prefetch((const void*)(featb + (size_t)c * PP + pbase + TP + (tid >> 7) * 64), 0, 1);
        }
        __syncthreads();

        const int prow = wv * 16;   // this wave's 16 pixels within the tile
        if (lane < 16) atomicAdd(&segcnt[lbl[prow + lane]], 1.0f);

        // ---- GEMM1: h1[p][o] = relu(X[p][c] * W1[o][c]^T + b1) ----
        #pragma unroll
        for (int mo = 0; mo < 4; ++mo) {
            v8f acc = {0.f, 0.f, 0.f, 0.f, 0.f, 0.f, 0.f, 0.f};
            #pragma unroll
            for (int kk = 0; kk < 4; ++kk) {
                const int k0 = kk * 32;
                Frag a, bf;
                const unsigned short* arow = &Xb[prow + n][k0 + khalfA];
                a.q[0] = *(const u32x4*)(arow);
                a.q[1] = *(const u32x4*)(arow + 16);
                const unsigned short* brow = &W1s[mo * 16 + n][k0 + khB];
                bf.q[0] = *(const u32x4*)(brow);
                bf.q[1] = *(const u32x4*)(brow + 16);
                acc = __builtin_amdgcn_wmma_f32_16x16x32_bf16(
                          false, a.v, false, bf.v, (short)0, acc, false, false);
            }
            const float bias = c1b[mo * 16 + n];
            #pragma unroll
            for (int v = 0; v < 8; ++v) {
                float h = acc[v] + bias;
                h = h > 0.0f ? h : 0.0f;
                H1s[wv][v + 8 * hi][mo * 16 + n] = f2bf(h);   // [p][d]
            }
        }
        // wave-private H1s: DS ops are in-order within a wave, no barrier needed

        // ---- GEMM2: feat[p][o2] = H1[p][d] * W2[o2][d]^T + b2, then segment add ----
        #pragma unroll
        for (int mo = 0; mo < 4; ++mo) {
            v8f acc = {0.f, 0.f, 0.f, 0.f, 0.f, 0.f, 0.f, 0.f};
            #pragma unroll
            for (int kk = 0; kk < 2; ++kk) {
                const int k0 = kk * 32;
                Frag a, bf;
                const unsigned short* arow = &H1s[wv][n][k0 + khalfA];
                a.q[0] = *(const u32x4*)(arow);
                a.q[1] = *(const u32x4*)(arow + 16);
                const unsigned short* brow = &W2s[mo * 16 + n][k0 + khB];
                bf.q[0] = *(const u32x4*)(brow);
                bf.q[1] = *(const u32x4*)(brow + 16);
                acc = __builtin_amdgcn_wmma_f32_16x16x32_bf16(
                          false, a.v, false, bf.v, (short)0, acc, false, false);
            }
            const float bias = c2b[mo * 16 + n];
            #pragma unroll
            for (int v = 0; v < 8; ++v) {
                const int p  = v + 8 * hi;
                const int sl = lbl[prow + p];
                atomicAdd(&segsum[sl][mo * 16 + n], acc[v] + bias);  // ds_add_f32
            }
        }
        __syncthreads();
    }

    // flush per-block partials to global accumulators
    for (int i = tid; i < SS * DD; i += 256)
        atomicAdd(&gsum[(size_t)b * SS * DD + i], segsum[i / DD][i % DD]);
    if (tid < SS) atomicAdd(&gcnt[b * SS + tid], segcnt[tid]);
}

__global__ __launch_bounds__(256)
void seg_finalize_kernel(const float* __restrict__ gsum, const float* __restrict__ gcnt,
                         const float* __restrict__ emb,
                         const float* __restrict__ ow, const float* __restrict__ ob,
                         float* __restrict__ out) {
    __shared__ float pooled[SS][DD];
    __shared__ int dst[SS];
    const int b = blockIdx.x;
    const int tid = threadIdx.x;

    for (int i = tid; i < SS * DD; i += 256) {
        int s = i / DD, d = i % DD;
        float c = gcnt[b * SS + s];
        pooled[s][d] = gsum[((size_t)b * SS + s) * DD + d] / fmaxf(c, 1.0f);
    }
    if (tid == 0) {
        int r = 0;
        for (int s = 0; s < SS; ++s) dst[s] = (gcnt[b * SS + s] > 0.0f) ? r++ : -1;
    }
    for (int i = tid; i < SS * DD; i += 256) out[(size_t)b * SS * DD + i] = 0.0f;
    __syncthreads();

    for (int i = tid; i < SS * DD; i += 256) {
        int s = i / DD, o = i % DD;
        if (dst[s] < 0) continue;
        float acc = ob[o];
        const float* wr = ow + (size_t)o * (2 * DD);
        #pragma unroll 4
        for (int e = 0; e < DD; ++e) acc += wr[e] * pooled[s][e];
        #pragma unroll 4
        for (int e = 0; e < DD; ++e) acc += wr[DD + e] * emb[s * DD + e];
        out[((size_t)b * SS + dst[s]) * DD + o] = acc;
    }
}

extern "C" void kernel_launch(void* const* d_in, const int* in_sizes, int n_in,
                              void* d_out, int out_size, void* d_ws, size_t ws_size,
                              hipStream_t stream) {
    (void)in_sizes; (void)n_in; (void)out_size; (void)ws_size;
    const int*   labels   = (const int*)d_in[0];
    const float* features = (const float*)d_in[1];
    const float* c1w      = (const float*)d_in[2];
    const float* c1b      = (const float*)d_in[3];
    const float* c2w      = (const float*)d_in[4];
    const float* c2b      = (const float*)d_in[5];
    const float* emb      = (const float*)d_in[6];
    const float* ow       = (const float*)d_in[7];
    const float* ob       = (const float*)d_in[8];
    float* out  = (float*)d_out;
    float* gsum = (float*)d_ws;                  // BN*SS*DD
    float* gcnt = gsum + BN * SS * DD;           // BN*SS

    const int nws = BN * SS * DD + BN * SS;
    seg_init_kernel<<<(nws + 255) / 256, 256, 0, stream>>>(gsum, nws);
    seg_fused_kernel<<<BN * BLK_PER_B, 256, 0, stream>>>(labels, features, c1w, c1b,
                                                         c2w, c2b, gsum, gcnt);
    seg_finalize_kernel<<<BN, 256, 0, stream>>>(gsum, gcnt, emb, ow, ob, out);
}
